// CrossAttention_8340826489080
// MI455X (gfx1250) — matrix operand order repaired and checked by reading, not compiled, not run on hardware
//
#include <hip/hip_runtime.h>
#include <hip/hip_bf16.h>

// ---------------------------------------------------------------------------
// CDNA5 (gfx1250) wave32 WMMA attention:
//   q = x Wq^T, k = x Wk^T, v = x Wv^T  (f16 in, f32 accumulate)
//   flash-style softmax(q k^T * scale) v, V tile staged to LDS with
//   GLOBAL_LOAD_ASYNC_TO_LDS_B128 (ASYNCcnt) to hide fetch latency
//   out = attn Wo^T + bo  (fp32 out)
// ---------------------------------------------------------------------------

typedef _Float16 h16;
typedef __attribute__((ext_vector_type(16))) _Float16 v16h;
typedef __attribute__((ext_vector_type(8)))  _Float16 v8h;
typedef __attribute__((ext_vector_type(8)))  float    v8f;

#define HEADS    16
#define DIM_HEAD 64
#define ATT_SCALE 0.125f   // 64^-0.5
#define BATCH    2
#define SEQ      2048
#define DMODEL   1024
#define ROWS     (BATCH * SEQ)      // 4096
#define NEG_BIG  (-3.0e38f)

union ABreg { v16h v; v8h h[2]; };

// A operand 16x32 (MxK), f16, row-major source with K contiguous.
// Layout (ISA 7.12.2): lanes 0-15 -> row=lr, K chunks {0..7, 16..23};
//                      lanes 16-31 -> row=lr, K chunks {8..15, 24..31}.
__device__ __forceinline__ v16h load_a_f16(const h16* base, int ld, int lane) {
  const int g = lane >> 4, lr = lane & 15;
  const h16* row = base + (size_t)lr * ld;
  ABreg a;
  a.h[0] = *(const v8h*)(row + g * 8);
  a.h[1] = *(const v8h*)(row + 16 + g * 8);
  return a.v;
}

// B operand 32x16 (KxN) taken from a row-major [N, K] matrix (K contiguous).
// Layout: lanes 0-15 -> col N=lr, K chunks {0..7, 16..23}; lanes 16-31 -> col N=lr, K chunks {8..15, 24..31}.
__device__ __forceinline__ v16h load_b_f16(const h16* base, int ld, int lane) {
  const int g = lane >> 4, lr = lane & 15;
  const h16* row = base + (size_t)lr * ld;
  ABreg b;
  b.h[0] = *(const v8h*)(row + g * 8);
  b.h[1] = *(const v8h*)(row + 16 + g * 8);
  return b.v;
}

__device__ __forceinline__ v8f wmma_f16(v16h a, v16h b, v8f c) {
  // (neg_a, A, neg_b, B, c_mod, C, reuse_a, reuse_b)
  return __builtin_amdgcn_wmma_f32_16x16x32_f16(false, a, false, b, (short)0, c,
                                                false, false);
}

// Async global->LDS copy, 16B per lane, tracked by ASYNCcnt.
__device__ __forceinline__ void async_to_lds_b128(unsigned lds_byte,
                                                  const void* gptr) {
  asm volatile("global_load_async_to_lds_b128 %0, %1, off"
               :: "v"(lds_byte), "v"((unsigned long long)(size_t)gptr)
               : "memory");
}
__device__ __forceinline__ void wait_asynccnt0() {
  asm volatile("s_wait_asynccnt 0x0" ::: "memory");
}

// ---------------------------------------------------------------------------
__global__ __launch_bounds__(256) void f32_to_f16_kernel(const float* __restrict__ in,
                                                         h16* __restrict__ out, int n) {
  int i = blockIdx.x * blockDim.x + threadIdx.x;
  if (i < n) out[i] = (h16)in[i];
}

// ---------------------------------------------------------------------------
// Fused QKV projection. One wave computes a 16x64 tile of Q, K and V.
// Q,K stored [ROWS, DMODEL] f16; V stored transposed [B, H, Dh, S] f16.
__global__ __launch_bounds__(256) void qkv_gemm_kernel(
    const h16* __restrict__ xh,
    const h16* __restrict__ wq, const h16* __restrict__ wk, const h16* __restrict__ wv,
    h16* __restrict__ qh, h16* __restrict__ kh, h16* __restrict__ vth) {
  const int tid = threadIdx.x, wid = tid >> 5, lane = tid & 31;
  const int g = lane >> 4, lr = lane & 15;
  const int wg = blockIdx.x * 8 + wid;          // 4096 waves
  const int mt = wg >> 4;                       // 0..255  (16-row tiles)
  const int nt = wg & 15;                       // 0..15   (64-col tiles == head)
  const int m0 = mt * 16, n0 = nt * 64;

  v8f aq[4] = {}, ak[4] = {}, av[4] = {};

  for (int ks = 0; ks < DMODEL / 32; ++ks) {
    const int k0 = ks * 32;
    const v16h a = load_a_f16(xh + (size_t)m0 * DMODEL + k0, DMODEL, lane);
#pragma unroll
    for (int ct = 0; ct < 4; ++ct) {
      const size_t woff = (size_t)(n0 + ct * 16) * DMODEL + k0;
      aq[ct] = wmma_f16(a, load_b_f16(wq + woff, DMODEL, lane), aq[ct]);
      ak[ct] = wmma_f16(a, load_b_f16(wk + woff, DMODEL, lane), ak[ct]);
      av[ct] = wmma_f16(a, load_b_f16(wv + woff, DMODEL, lane), av[ct]);
    }
  }

#pragma unroll
  for (int ct = 0; ct < 4; ++ct) {
#pragma unroll
    for (int r = 0; r < 8; ++r) {
      const int row = m0 + 8 * g + r;           // global row in [0,4096)
      const int col = n0 + ct * 16 + lr;        // feature in [0,1024)
      qh[(size_t)row * DMODEL + col] = (h16)aq[ct][r];
      kh[(size_t)row * DMODEL + col] = (h16)ak[ct][r];
      const int b = row >> 11, s = row & (SEQ - 1);
      const int h = col >> 6, d = col & 63;     // head, dim-in-head
      vth[(((size_t)b * HEADS + h) * DIM_HEAD + d) * SEQ + s] = (h16)av[ct][r];
    }
  }
}

// ---------------------------------------------------------------------------
// Flash attention: one wave owns a 16-query tile of one (b,h); iterates the
// 2048 keys in chunks of 32 with online softmax. V^T tile (64x32 f16, 4KB)
// is staged into LDS with async loads and consumed after s_wait_asynccnt.
__global__ __launch_bounds__(256) void flash_attn_kernel(
    const h16* __restrict__ qh, const h16* __restrict__ kh,
    const h16* __restrict__ vth, h16* __restrict__ attnh) {
  __shared__ __align__(32) h16 plds[8 * 16 * 32];   // 1KB/wave: P tile C->A relayout
  __shared__ __align__(32) h16 vlds[8 * 64 * 32];   // 4KB/wave: staged V^T tile

  const int tid = threadIdx.x, wid = tid >> 5, lane = tid & 31;
  const int g = lane >> 4, lr = lane & 15;
  const int wg = blockIdx.x * 8 + wid;          // 4096 waves
  const int bh = wg >> 7;                       // 0..31 = b*16+h
  const int qt = wg & 127;                      // 0..127
  const int b = bh >> 4, h = bh & 15;
  const int qbase = qt * 16;

  const h16* qptr  = qh + ((size_t)b * SEQ + qbase) * DMODEL + h * DIM_HEAD;
  const h16* krow0 = kh + (size_t)b * SEQ * DMODEL + h * DIM_HEAD;
  const h16* vbase = vth + (size_t)bh * DIM_HEAD * SEQ;    // [Dh][S] for this head
  h16* myp = plds + wid * (16 * 32);
  h16* myv = vlds + wid * (64 * 32);
  const unsigned myv_byte = (unsigned)(size_t)myv;         // flat addr low 32b = LDS addr
  // async lane mapping: 4 lanes per V^T row, 16B each
  const int vd_lane = lane >> 2;        // dim sub-index 0..7
  const int vc_lane = (lane & 3) * 8;   // half offset within 32-key row

  const v16h qa0 = load_a_f16(qptr, DMODEL, lane);        // features  0..31
  const v16h qa1 = load_a_f16(qptr + 32, DMODEL, lane);   // features 32..63

  v8f o[4] = {};
  float mi[8], li[8];
#pragma unroll
  for (int r = 0; r < 8; ++r) { mi[r] = NEG_BIG; li[r] = 0.0f; }

  for (int kt = 0; kt < SEQ / 32; ++kt) {
    const int kbase = kt * 32;

    // Kick off async V^T tile fetch: 64 dims x 32 keys -> LDS [dim][key].
#pragma unroll
    for (int i = 0; i < 8; ++i) {
      const int d = i * 8 + vd_lane;                       // 0..63
      async_to_lds_b128(myv_byte + (unsigned)(d * 64 + vc_lane * 2),
                        vbase + (size_t)d * SEQ + kbase + vc_lane);
    }

    // Scores while the V tile is in flight.
    const h16* kb  = krow0 + (size_t)kbase * DMODEL;      // keys kbase..kbase+15
    const h16* kb2 = kb + (size_t)16 * DMODEL;            // keys kbase+16..+31
    v8f s0 = {}, s1 = {};
    s0 = wmma_f16(qa0, load_b_f16(kb,       DMODEL, lane), s0);
    s0 = wmma_f16(qa1, load_b_f16(kb  + 32, DMODEL, lane), s0);
    s1 = wmma_f16(qa0, load_b_f16(kb2,      DMODEL, lane), s1);
    s1 = wmma_f16(qa1, load_b_f16(kb2 + 32, DMODEL, lane), s1);

    // Online softmax; per lane: 8 rows (M = 8g + r), column N = lr / 16+lr.
#pragma unroll
    for (int r = 0; r < 8; ++r) {
      float x0 = s0[r] * ATT_SCALE;
      float x1 = s1[r] * ATT_SCALE;
      float rm = fmaxf(x0, x1);
      rm = fmaxf(rm, __shfl_xor(rm, 1, 32));
      rm = fmaxf(rm, __shfl_xor(rm, 2, 32));
      rm = fmaxf(rm, __shfl_xor(rm, 4, 32));
      rm = fmaxf(rm, __shfl_xor(rm, 8, 32));
      const float nm  = fmaxf(mi[r], rm);
      const float fac = __expf(mi[r] - nm);
      const float p0  = __expf(x0 - nm);
      const float p1  = __expf(x1 - nm);
      float ps = p0 + p1;
      ps += __shfl_xor(ps, 1, 32);
      ps += __shfl_xor(ps, 2, 32);
      ps += __shfl_xor(ps, 4, 32);
      ps += __shfl_xor(ps, 8, 32);
      li[r] = li[r] * fac + ps;
      mi[r] = nm;
      o[0][r] *= fac; o[1][r] *= fac; o[2][r] *= fac; o[3][r] *= fac;
      // P tile 16x32 row-major into LDS (row = query, col = local key)
      myp[(8 * g + r) * 32 + lr]      = (h16)p0;
      myp[(8 * g + r) * 32 + 16 + lr] = (h16)p1;
    }
    __syncthreads();
    wait_asynccnt0();                                     // V tile resident in LDS

    const v16h pa = load_a_f16(myp, 32, lane);            // P as A operand (K=32 keys)
#pragma unroll
    for (int ct = 0; ct < 4; ++ct) {
      // B operand from staged LDS tile: rows = dims, 32 local keys contiguous.
      o[ct] = wmma_f16(pa, load_b_f16(myv + (size_t)(ct * 16) * 32, 32, lane), o[ct]);
    }
    __syncthreads();
  }

#pragma unroll
  for (int r = 0; r < 8; ++r) {
    const float inv = 1.0f / li[r];
    const int row = qbase + 8 * g + r;
    h16* op = attnh + ((size_t)b * SEQ + row) * DMODEL + h * DIM_HEAD + lr;
    op[0]  = (h16)(o[0][r] * inv);
    op[16] = (h16)(o[1][r] * inv);
    op[32] = (h16)(o[2][r] * inv);
    op[48] = (h16)(o[3][r] * inv);
  }
}

// ---------------------------------------------------------------------------
// Output projection: out = attn @ Wo^T + bo, fp32 result.
__global__ __launch_bounds__(256) void out_proj_kernel(
    const h16* __restrict__ ah, const h16* __restrict__ wo,
    const float* __restrict__ bo, float* __restrict__ out) {
  const int tid = threadIdx.x, wid = tid >> 5, lane = tid & 31;
  const int g = lane >> 4, lr = lane & 15;
  const int wg = blockIdx.x * 8 + wid;
  const int mt = wg >> 4, nt = wg & 15;
  const int m0 = mt * 16, n0 = nt * 64;

  v8f acc[4] = {};
  for (int ks = 0; ks < DMODEL / 32; ++ks) {
    const int k0 = ks * 32;
    const v16h a = load_a_f16(ah + (size_t)m0 * DMODEL + k0, DMODEL, lane);
#pragma unroll
    for (int ct = 0; ct < 4; ++ct) {
      const size_t woff = (size_t)(n0 + ct * 16) * DMODEL + k0;
      acc[ct] = wmma_f16(a, load_b_f16(wo + woff, DMODEL, lane), acc[ct]);
    }
  }
#pragma unroll
  for (int ct = 0; ct < 4; ++ct) {
    const int col = n0 + ct * 16 + lr;
    const float bias = bo[col];
#pragma unroll
    for (int r = 0; r < 8; ++r) {
      const int row = m0 + 8 * g + r;
      out[(size_t)row * DMODEL + col] = acc[ct][r] + bias;
    }
  }
}

// ---------------------------------------------------------------------------
extern "C" void kernel_launch(void* const* d_in, const int* in_sizes, int n_in,
                              void* d_out, int out_size, void* d_ws, size_t ws_size,
                              hipStream_t stream) {
  (void)in_sizes; (void)n_in; (void)out_size; (void)ws_size;
  const float* x  = (const float*)d_in[0];
  const float* Wq = (const float*)d_in[1];
  const float* Wk = (const float*)d_in[2];
  const float* Wv = (const float*)d_in[3];
  const float* Wo = (const float*)d_in[4];
  const float* bo = (const float*)d_in[5];
  float* out = (float*)d_out;

  // Workspace carving (f16): ~48 MB total, fits L2-resident working set.
  h16* xh    = (h16*)d_ws;                         // 4096*1024
  h16* wqh   = xh  + (size_t)ROWS * DMODEL;        // 1024*1024
  h16* wkh   = wqh + (size_t)DMODEL * DMODEL;
  h16* wvh   = wkh + (size_t)DMODEL * DMODEL;
  h16* woh   = wvh + (size_t)DMODEL * DMODEL;
  h16* qh    = woh + (size_t)DMODEL * DMODEL;      // 4096*1024
  h16* kh    = qh  + (size_t)ROWS * DMODEL;
  h16* vth   = kh  + (size_t)ROWS * DMODEL;        // [B,H,Dh,S]
  h16* attnh = vth + (size_t)ROWS * DMODEL;

  const int nx = ROWS * DMODEL;        // 4,194,304
  const int nw = DMODEL * DMODEL;      // 1,048,576
  dim3 blk(256);
  f32_to_f16_kernel<<<(nx + 255) / 256, blk, 0, stream>>>(x,  xh,  nx);
  f32_to_f16_kernel<<<(nw + 255) / 256, blk, 0, stream>>>(Wq, wqh, nw);
  f32_to_f16_kernel<<<(nw + 255) / 256, blk, 0, stream>>>(Wk, wkh, nw);
  f32_to_f16_kernel<<<(nw + 255) / 256, blk, 0, stream>>>(Wv, wvh, nw);
  f32_to_f16_kernel<<<(nw + 255) / 256, blk, 0, stream>>>(Wo, woh, nw);

  // 4096 waves each: (ROWS/16) * (DMODEL/64) tiles, 8 waves / 256-thread block.
  qkv_gemm_kernel<<<512, blk, 0, stream>>>(xh, wqh, wkh, wvh, qh, kh, vth);
  // 4096 waves: B*H * (SEQ/16) query tiles.
  flash_attn_kernel<<<512, blk, 0, stream>>>(qh, kh, vth, attnh);
  out_proj_kernel<<<512, blk, 0, stream>>>(attnh, woh, bo, out);
}
